// Tail_29824252903889
// MI455X (gfx1250) — matrix operand order repaired and checked by reading, not compiled, not run on hardware
//
#include <hip/hip_runtime.h>
#include <hip/hip_bf16.h>

// ---------------------------------------------------------------------------
// Problem constants (from reference)
// ---------------------------------------------------------------------------
constexpr int B_   = 32;
constexpr int T_   = 16;
constexpr int M_   = B_ * T_;      // 512
constexpr int N_   = 1024;         // DM
constexpr int GK   = 2048 * 49;    // 100352
constexpr int NC   = 1000;

// GEMM tiling
constexpr int BK      = 32;
constexpr int SPLITK  = 8;
constexpr int KSLICE  = GK / SPLITK;   // 12544
constexpr int KSTEPS  = KSLICE / BK;   // 392
constexpr int PA      = 40;            // LDS row pitch in halves (80B, 16B-aligned)
constexpr int BIASP   = 2052;          // bias table row pitch (padded)

#define SQRT_NF 45.254833995939045f    // sqrt(2048)

typedef __attribute__((ext_vector_type(16))) __bf16 v16bf;
typedef __attribute__((ext_vector_type(8)))  float  v8f;

union Frag { uint4 q[2]; v16bf v; };
union H4   { unsigned short h[4]; uint2 u2; };

__device__ inline unsigned short f2bf(float f) {   // RNE float -> bf16
    unsigned u = __float_as_uint(f);
    return (unsigned short)((u + 0x7FFFu + ((u >> 16) & 1u)) >> 16);
}

__device__ inline float blockReduceSum(float v, float* red) {
    for (int o = 16; o; o >>= 1) v += __shfl_xor(v, o, 32);
    int wid = threadIdx.x >> 5, lane = threadIdx.x & 31;
    __syncthreads();
    if (lane == 0) red[wid] = v;
    __syncthreads();
    float t = 0.f;
    int nw = blockDim.x >> 5;
    for (int i = 0; i < nw; i++) t += red[i];
    __syncthreads();
    return t;
}

// ---------------------------------------------------------------------------
// Kernel 0: precompute fused BN+PE affine tables
//   scaleT[c] = gamma/sqrt(var+1e-5) * sqrt(NF)
//   biasT[f,c]= (beta - mean*gamma/sqrt(var+1e-5))*sqrt(NF) + pe[f,c]
// pe: even col -> sin(f * 10000^(-2c/d)), odd col -> cos(same)   (matches ref)
// ---------------------------------------------------------------------------
__global__ void pre_kernel(const float* __restrict__ g, const float* __restrict__ bt,
                           const float* __restrict__ mn, const float* __restrict__ vr,
                           float* __restrict__ scaleT, float* __restrict__ biasT) {
    int c = blockIdx.x * 256 + threadIdx.x;
    if (c >= BIASP) return;
    if (c >= 2048) {                       // zero pad (c0+1 overreads land here)
        scaleT[c] = 0.f;
        for (int f = 0; f < 16; f++) biasT[f * BIASP + c] = 0.f;
        return;
    }
    float inv = g[c] / sqrtf(vr[c] + 1e-5f);
    scaleT[c] = inv * SQRT_NF;
    float add = (bt[c] - mn[c] * inv) * SQRT_NF;
    float freq = powf(10000.f, -2.f * (float)c / 2048.f);
    for (int f = 0; f < 16; f++) {
        float arg = (float)f * freq;
        float pe  = (c & 1) ? cosf(arg) : sinf(arg);
        biasT[f * BIASP + c] = add + pe;
    }
}

// ---------------------------------------------------------------------------
// Kernel 1: fused BN+PE + conv-as-GEMM, bf16 WMMA, split-K partials
//   Cp[s, m, n] = sum_{k in slice s} (x[m,k]*scale[c(k)] + bias[m%16,c(k)]) * Wq[n,k]
// ---------------------------------------------------------------------------
__global__ __launch_bounds__(256) void gemm_kernel(
    const float* __restrict__ X, const float* __restrict__ Wq,
    const float* __restrict__ scaleT, const float* __restrict__ biasT,
    float* __restrict__ Cp)
{
    __shared__ __align__(16) unsigned short ldsA[2][128 * PA];
    __shared__ __align__(16) unsigned short ldsB[2][128 * PA];

    const int tid  = threadIdx.x;
    const int lane = tid & 31;
    const int wid  = tid >> 5;
    const int wm   = (wid & 3) * 32;     // wave 32x64 sub-tile
    const int wn   = (wid >> 2) * 64;
    const int m0   = (blockIdx.x >> 3) * 128;
    const int n0   = (blockIdx.x & 7) * 128;
    const int k00  = blockIdx.y * KSLICE;

    v8f acc[2][4] = {};

    // load slots: 128 rows x 8 float4 = 1024 slots / 256 threads = 4 each
    int rowA[4], kqv[4];
    for (int i = 0; i < 4; i++) {
        int slot = tid + i * 256;
        rowA[i] = slot >> 3;
        kqv[i]  = (slot & 7) * 4;
    }

    float4 ra[4], rb[4];
    float  bias0[4], bias1[4];
    float  cs0 = 0.f, cs1 = 0.f;
    int    cbnd = 0, kcur = 0;

    auto LOAD = [&](int kb) {
        int c0 = kb / 49;                    // window spans <= 2 channels (49 > 32)
        cs0 = scaleT[c0]; cs1 = scaleT[c0 + 1];
        cbnd = (c0 + 1) * 49;
        kcur = kb;
        for (int i = 0; i < 4; i++) {
            ra[i] = *(const float4*)(X  + (size_t)(m0 + rowA[i]) * GK + kb + kqv[i]);
            rb[i] = *(const float4*)(Wq + (size_t)(n0 + rowA[i]) * GK + kb + kqv[i]);
            int f = (m0 + rowA[i]) & 15;     // frame index
            bias0[i] = biasT[f * BIASP + c0];
            bias1[i] = biasT[f * BIASP + c0 + 1];
        }
    };
    auto STORE = [&](int buf) {
        for (int i = 0; i < 4; i++) {
            H4 ha, hb;
            float av[4] = { ra[i].x, ra[i].y, ra[i].z, ra[i].w };
            float bv[4] = { rb[i].x, rb[i].y, rb[i].z, rb[i].w };
            for (int j = 0; j < 4; j++) {
                int  kg = kcur + kqv[i] + j;
                bool hi2 = (kg >= cbnd);
                float s  = hi2 ? cs1 : cs0;
                float bb = hi2 ? bias1[i] : bias0[i];
                ha.h[j] = f2bf(fmaf(av[j], s, bb));
                hb.h[j] = f2bf(bv[j]);
            }
            *(uint2*)&ldsA[buf][rowA[i] * PA + kqv[i]] = ha.u2;
            *(uint2*)&ldsB[buf][rowA[i] * PA + kqv[i]] = hb.u2;
        }
    };

    const int r  = lane & 15;    // fragment row/col within 16
    const int hi = lane >> 4;    // lane half -> K group

    LOAD(k00);
    for (int step = 0; step < KSTEPS; ++step) {
        int buf = step & 1;
        STORE(buf);                                       // regs from prev LOAD
        if (step + 1 < KSTEPS) LOAD(k00 + (step + 1) * BK); // prefetch next
        __syncthreads();

        Frag af[2], bf2[4];
        for (int mi = 0; mi < 2; mi++) {
            // A 16x32 bf16 layout: lanes<16 K=0..7,16..23 ; lanes>=16 K=8..15,24..31
            const unsigned short* p = &ldsA[buf][(wm + mi * 16 + r) * PA + hi * 8];
            af[mi].q[0] = *(const uint4*)p;
            af[mi].q[1] = *(const uint4*)(p + 16);
        }
        for (int ni = 0; ni < 4; ni++) {
            // B 32x16: lane holds one column; lanes<16 K=0..7,16..23 ; lanes>=16 K=8..15,24..31
            const unsigned short* p = &ldsB[buf][(wn + ni * 16 + r) * PA + hi * 8];
            bf2[ni].q[0] = *(const uint4*)p;
            bf2[ni].q[1] = *(const uint4*)(p + 16);
        }
        for (int mi = 0; mi < 2; mi++)
            for (int ni = 0; ni < 4; ni++)
                acc[mi][ni] = __builtin_amdgcn_wmma_f32_16x16x32_bf16(
                    false, af[mi].v, false, bf2[ni].v, (short)0, acc[mi][ni],
                    false, false);
        __syncthreads();
    }

    // deterministic split-K: each slice writes its own partial buffer
    float* Cs = Cp + (size_t)blockIdx.y * M_ * N_;
    for (int mi = 0; mi < 2; mi++)
        for (int ni = 0; ni < 4; ni++) {
            int colg  = n0 + wn + ni * 16 + r;
            int rbase = m0 + wm + mi * 16 + hi * 8;   // C layout: VGPR j -> M=j(+8)
            for (int e = 0; e < 8; e++)
                Cs[(size_t)(rbase + e) * N_ + colg] = acc[mi][ni][e];
        }
}

// ---------------------------------------------------------------------------
// Kernel 2: split-K reduce (fixed order) + ReLU + row-norm (ddof=1, eps on std)
// ---------------------------------------------------------------------------
__global__ __launch_bounds__(256) void relunorm_kernel(const float* __restrict__ Cp,
                                                       float* __restrict__ xqn) {
    int row = blockIdx.x;                 // 512 rows
    int tid = threadIdx.x;                // 256 threads, 4 cols each
    __shared__ float red[8];
    float v[4], s = 0.f;
    for (int i = 0; i < 4; i++) {
        int col = tid + i * 256;
        float a = 0.f;
        for (int sl = 0; sl < SPLITK; sl++)
            a += Cp[((size_t)sl * M_ + row) * N_ + col];
        a = fmaxf(a, 0.f);
        v[i] = a; s += a;
    }
    float mu = blockReduceSum(s, red) * (1.f / 1024.f);
    float q2 = 0.f;
    for (int i = 0; i < 4; i++) { float d = v[i] - mu; q2 += d * d; }
    float sd  = sqrtf(blockReduceSum(q2, red) * (1.f / 1023.f));
    float inv = 1.f / (sd + 1e-6f);
    for (int i = 0; i < 4; i++)
        xqn[(size_t)row * N_ + tid + i * 256] = (v[i] - mu) * inv;
}

// ---------------------------------------------------------------------------
// Kernel 3: 3-layer single-query attention + FFN per (b, h). 64 threads/block.
// ---------------------------------------------------------------------------
__device__ inline float norm64(float u, float* tmp, int d) {
    __syncthreads();
    tmp[d] = u;
    __syncthreads();
    float mu = 0.f;
    for (int j = 0; j < 64; j++) mu += tmp[j];
    mu *= (1.f / 64.f);
    float q2 = 0.f;
    for (int j = 0; j < 64; j++) { float dd = tmp[j] - mu; q2 += dd * dd; }
    float sd = sqrtf(q2 * (1.f / 63.f));
    return (u - mu) / (sd + 1e-6f);
}

__global__ __launch_bounds__(64) void attn_kernel(
    const float* __restrict__ xqn,
    const float* __restrict__ w1, const float* __restrict__ b1,
    const float* __restrict__ w2, const float* __restrict__ b2,
    float* __restrict__ fbuf)
{
    int blk = blockIdx.x;
    int b = blk >> 4, h = blk & 15;
    int d = threadIdx.x;                    // 0..63
    __shared__ float kv[16][68];
    __shared__ float qv[64], tmp[64], hm[32], sc[16];

    for (int t = 0; t < 16; t++)
        kv[t][d] = xqn[(size_t)(b * 16 + t) * 1024 + h * 64 + d];
    float q = xqn[(size_t)(b * 16 + 8) * 1024 + h * 64 + d];   // t//2
    __syncthreads();

    for (int l = 0; l < 3; l++) {
        qv[d] = q;
        __syncthreads();
        if (d < 16) {
            float s = 0.f;
            for (int j = 0; j < 64; j++) s += qv[j] * kv[d][j];
            sc[d] = s * 0.125f;                                // 1/sqrt(64)
        }
        __syncthreads();
        float mx = sc[0];
        for (int t = 1; t < 16; t++) mx = fmaxf(mx, sc[t]);
        float ex[16], den = 0.f;
        for (int t = 0; t < 16; t++) { ex[t] = expf(sc[t] - mx); den += ex[t]; }
        float ida = 1.f / den;
        float A = 0.f;
        for (int t = 0; t < 16; t++) A += ex[t] * ida * kv[t][d];

        float u = norm64(A + q, tmp, d);
        __syncthreads();
        qv[d] = u;
        __syncthreads();
        if (d < 32) {
            const float* w1p = w1 + ((size_t)(l * 16 + h) * 32 + d) * 64;
            float s = b1[(l * 16 + h) * 32 + d];
            for (int j = 0; j < 64; j++) s += qv[j] * w1p[j];
            hm[d] = fmaxf(s, 0.f);
        }
        __syncthreads();
        const float* w2p = w2 + ((size_t)(l * 16 + h) * 64 + d) * 32;
        float f2 = b2[(l * 16 + h) * 64 + d];
        for (int j = 0; j < 32; j++) f2 += hm[j] * w2p[j];
        q = norm64(u + f2, tmp, d);
        __syncthreads();
    }
    fbuf[(size_t)b * 1024 + h * 64 + d] = q;
}

// ---------------------------------------------------------------------------
// Kernel 4: L2-normalize f and classifier  out = (f/||f||) @ Wc^T + bc
// ---------------------------------------------------------------------------
__global__ __launch_bounds__(256) void head_kernel(
    const float* __restrict__ fbuf, const float* __restrict__ Wc,
    const float* __restrict__ bc, float* __restrict__ out)
{
    int b = blockIdx.x, tid = threadIdx.x;
    __shared__ float fr[1024];
    __shared__ float red[8];
    float ss = 0.f;
    for (int i = 0; i < 4; i++) {
        float v = fbuf[(size_t)b * 1024 + tid + i * 256];
        fr[tid + i * 256] = v;
        ss += v * v;
    }
    float inv = 1.f / fmaxf(sqrtf(blockReduceSum(ss, red)), 1e-12f);
    for (int c = tid; c < NC; c += 256) {
        const float* w = Wc + (size_t)c * 1024;
        float s = 0.f;
        for (int j = 0; j < 1024; j++) s += fr[j] * w[j];
        out[b * NC + c] = s * inv + bc[c];
    }
}

// ---------------------------------------------------------------------------
extern "C" void kernel_launch(void* const* d_in, const int* in_sizes, int n_in,
                              void* d_out, int out_size, void* d_ws, size_t ws_size,
                              hipStream_t stream)
{
    (void)in_sizes; (void)n_in; (void)out_size; (void)ws_size;
    const float* x     = (const float*)d_in[0];
    const float* gamma = (const float*)d_in[1];
    const float* beta  = (const float*)d_in[2];
    const float* mean  = (const float*)d_in[3];
    const float* var   = (const float*)d_in[4];
    const float* Wq    = (const float*)d_in[5];
    const float* w1    = (const float*)d_in[6];
    const float* b1    = (const float*)d_in[7];
    const float* w2    = (const float*)d_in[8];
    const float* b2    = (const float*)d_in[9];
    const float* Wc    = (const float*)d_in[10];
    const float* bc    = (const float*)d_in[11];

    float* ws     = (float*)d_ws;
    float* scaleT = ws;                         //  4096
    float* biasT  = ws + 4096;                  // 16*2052 -> reserve 36864
    float* Cp     = ws + 40960;                 // SPLITK*512*1024 = 4194304
    float* xqn    = Cp + (size_t)SPLITK * M_ * N_;  // 524288
    float* fbuf   = xqn + (size_t)M_ * N_;      // 32768   (~19.2 MB total)

    pre_kernel     <<<9,            256, 0, stream>>>(gamma, beta, mean, var, scaleT, biasT);
    gemm_kernel    <<<dim3(32, SPLITK), 256, 0, stream>>>(x, Wq, scaleT, biasT, Cp);
    relunorm_kernel<<<M_,           256, 0, stream>>>(Cp, xqn);
    attn_kernel    <<<M_,            64, 0, stream>>>(xqn, w1, b1, w2, b2, fbuf);
    head_kernel    <<<B_,           256, 0, stream>>>(fbuf, Wc, bc, (float*)d_out);
}
